// LoRAAttention_51891794870864
// MI455X (gfx1250) — matrix operand order repaired and checked by reading, not compiled, not run on hardware
//
#include <hip/hip_runtime.h>

typedef unsigned short u16;
typedef unsigned int   u32;

typedef __attribute__((ext_vector_type(16))) __bf16 v16bf;
typedef __attribute__((ext_vector_type(8)))  float  v8f;

union Frag { v16bf v; u32 u[8]; };

static constexpr int    Bsz = 4, Nn = 2048, Cc = 768, Hh = 12, HD = 64;
static constexpr size_t HS  = (size_t)Bsz * Hh * Nn * HD;   // elems per q/k/v plane
static constexpr float  SCALING = 2.0f;    // ALPHA/RANK = 16/8
static constexpr float  QSCALE  = 0.125f;  // HD^-0.5
static constexpr float  LOG2E   = 1.44269504088896f;

// ---- CDNA5 async global->LDS path (ASYNCcnt), with safe fallback ----------
#if defined(__has_builtin)
#if __has_builtin(__builtin_amdgcn_global_load_async_to_lds_b128)
#define HAVE_ASYNC 1
#endif
#endif
#ifndef HAVE_ASYNC
#define HAVE_ASYNC 0
#endif

// builtin signature: (AS1 vi4* gsrc, AS3 vi4* ldst, int offset, int cpol)
typedef __attribute__((__vector_size__(16))) int vi4;
typedef __attribute__((address_space(1))) vi4* gvi4_ptr;
typedef __attribute__((address_space(3))) vi4* lvi4_ptr;
#define AS1(p) ((gvi4_ptr)(unsigned long long)(p))
#define AS3(p) ((lvi4_ptr)(u32)(unsigned long long)(p))

#if defined(__has_builtin) && __has_builtin(__builtin_amdgcn_s_wait_asynccnt)
#define ASYNC_WAIT() __builtin_amdgcn_s_wait_asynccnt(0)
#else
#define ASYNC_WAIT() asm volatile("s_wait_asynccnt 0x0" ::: "memory")
#endif

__device__ __forceinline__ u16 f2bf(float f) {
    u32 x = __float_as_uint(f);
    u32 r = x + 0x7FFFu + ((x >> 16) & 1u);
    return (u16)(r >> 16);
}

// Butterfly reductions over a 16-lane row via DPP16 ROW_XMASK (pure VALU).
template <int M>
__device__ __forceinline__ float dpp_xor(float x) {
    return __int_as_float(__builtin_amdgcn_update_dpp(
        0, __float_as_int(x), 0x160 + M, 0xF, 0xF, true));
}
__device__ __forceinline__ float rmax16(float v) {
    v = fmaxf(v, dpp_xor<1>(v));
    v = fmaxf(v, dpp_xor<2>(v));
    v = fmaxf(v, dpp_xor<4>(v));
    v = fmaxf(v, dpp_xor<8>(v));
    return v;
}
__device__ __forceinline__ float rsum16(float v) {
    v += dpp_xor<1>(v);
    v += dpp_xor<2>(v);
    v += dpp_xor<4>(v);
    v += dpp_xor<8>(v);
    return v;
}

// ---------------------------------------------------------------------------
// t[row][r] = sum_c X[row][c] * A[r][c]   (X: [8192][768] f32, A: [8][768])
// ---------------------------------------------------------------------------
__global__ __launch_bounds__(256) void lora_t_kernel(const float* __restrict__ X,
                                                     const float* __restrict__ A,
                                                     float* __restrict__ T) {
    int idx = blockIdx.x * 256 + threadIdx.x;      // 65536 = 8192*8
    int row = idx >> 3, r = idx & 7;
    const float4* xp = (const float4*)(X + (size_t)row * Cc);
    const float4* ap = (const float4*)(A + (size_t)r   * Cc);
    float s = 0.f;
    for (int i = 0; i < Cc / 4; ++i) {
        float4 a = xp[i], b = ap[i];
        s += a.x * b.x + a.y * b.y + a.z * b.z + a.w * b.w;
    }
    T[idx] = s;
}

// ---------------------------------------------------------------------------
// WMMA GEMM: out[M=8192][N] = Ain[M][768] @ W[N][768]^T, bf16 MACs, f32 acc.
// EPI==0 : qkv epilogue (bias pattern + LoRA + q-scale, scatter bf16 heads)
// EPI==1 : proj epilogue (+bias, store f32)
// Block tile 128x64, 8 waves of 32x32 (2x2 WMMA 16x16 tiles), K step 32.
// ---------------------------------------------------------------------------
template <bool AF32, int EPI>
__global__ __launch_bounds__(256) void gemm_kernel(const void*  __restrict__ Ain,
                                                   const float* __restrict__ W,
                                                   const float* __restrict__ b0,
                                                   const float* __restrict__ b1,
                                                   const float* __restrict__ T,
                                                   const float* __restrict__ LB,
                                                   u16*   __restrict__ obf,
                                                   float* __restrict__ of32) {
    __shared__ alignas(16) u16 As[128][40];
    __shared__ alignas(16) u16 Bs[64][40];

    const int tid = threadIdx.x, lane = tid & 31, wid = tid >> 5;
    const int wm = (wid & 3) * 32, wn = (wid >> 2) * 32;
    const int m0 = blockIdx.y * 128, n0 = blockIdx.x * 64;
    const int lm = lane & 15, hi = lane >> 4;

    v8f c00 = {}, c01 = {}, c10 = {}, c11 = {};

    for (int kc = 0; kc < Cc; kc += 32) {
        // ---- stage A tile (128x32) into LDS as bf16 ----
        if (AF32) {
            const float* A = (const float*)Ain;
            int r = tid >> 1, sg = (tid & 1) * 16;
            const float4* src = (const float4*)(A + (size_t)(m0 + r) * Cc + kc + sg);
            #pragma unroll
            for (int q = 0; q < 4; ++q) {
                float4 f = src[q];
                *(u32*)&As[r][sg + 4 * q]     = (u32)f2bf(f.x) | ((u32)f2bf(f.y) << 16);
                *(u32*)&As[r][sg + 4 * q + 2] = (u32)f2bf(f.z) | ((u32)f2bf(f.w) << 16);
            }
            if (kc + 32 < Cc) __builtin_prefetch(src + 8, 0, 3);
        } else {
            const u16* A = (const u16*)Ain;
            int r = tid >> 1, sg = (tid & 1) * 16;
            const u16* g = A + (size_t)(m0 + r) * Cc + kc + sg;
#if HAVE_ASYNC
            __builtin_amdgcn_global_load_async_to_lds_b128(AS1(g),     AS3(&As[r][sg]),     0, 0);
            __builtin_amdgcn_global_load_async_to_lds_b128(AS1(g + 8), AS3(&As[r][sg + 8]), 0, 0);
#else
            const uint4* src = (const uint4*)g;
            *(uint4*)&As[r][sg]     = src[0];
            *(uint4*)&As[r][sg + 8] = src[1];
#endif
        }
        // ---- stage B tile (64x32 of W) into LDS as bf16 ----
        {
            int r = tid >> 2, sg = (tid & 3) * 8;
            const float4* src = (const float4*)(W + (size_t)(n0 + r) * Cc + kc + sg);
            #pragma unroll
            for (int q = 0; q < 2; ++q) {
                float4 f = src[q];
                *(u32*)&Bs[r][sg + 4 * q]     = (u32)f2bf(f.x) | ((u32)f2bf(f.y) << 16);
                *(u32*)&Bs[r][sg + 4 * q + 2] = (u32)f2bf(f.z) | ((u32)f2bf(f.w) << 16);
            }
            if (kc + 32 < Cc) __builtin_prefetch(src + 8, 0, 3);
        }
#if HAVE_ASYNC
        if (!AF32) ASYNC_WAIT();
#endif
        __syncthreads();

        Frag a0, a1, bb0, bb1;
        #pragma unroll
        for (int j = 0; j < 8; ++j) {
            int ka = hi * 16 + 2 * j;   // A 16x32 layout
            a0.u[j]  = *(const u32*)&As[wm + lm][ka];
            a1.u[j]  = *(const u32*)&As[wm + 16 + lm][ka];
            int kb = hi * 16 + 2 * j;                               // B 32x16 layout
            bb0.u[j] = *(const u32*)&Bs[wn + lm][kb];
            bb1.u[j] = *(const u32*)&Bs[wn + 16 + lm][kb];
        }
        c00 = __builtin_amdgcn_wmma_f32_16x16x32_bf16(false, a0.v, false, bb0.v, (short)0, c00, false, false);
        c01 = __builtin_amdgcn_wmma_f32_16x16x32_bf16(false, a0.v, false, bb1.v, (short)0, c01, false, false);
        c10 = __builtin_amdgcn_wmma_f32_16x16x32_bf16(false, a1.v, false, bb0.v, (short)0, c10, false, false);
        c11 = __builtin_amdgcn_wmma_f32_16x16x32_bf16(false, a1.v, false, bb1.v, (short)0, c11, false, false);
        __syncthreads();
    }

    // ---- epilogue ----
    v8f* cc[4] = {&c00, &c01, &c10, &c11};
    #pragma unroll
    for (int t = 0; t < 4; ++t) {
        int mi = (t >> 1) * 16, ni = (t & 1) * 16;
        #pragma unroll
        for (int i = 0; i < 8; ++i) {
            int row = m0 + wm + mi + i + hi * 8;      // C layout: M = i (+8 hi-half)
            int col = n0 + wn + ni + lm;              //           N = lane%16
            float v = (*cc[t])[i];
            if (EPI == 0) {
                float bias = (col < Cc) ? b0[col] : ((col < 2 * Cc) ? 0.f : b1[col - 2 * Cc]);
                float s = 0.f;
                #pragma unroll
                for (int j = 0; j < 8; ++j) s += T[row * 8 + j] * LB[col * 8 + j];
                v += bias + SCALING * s;
                int three = col / Cc;
                int rem = col - three * Cc;
                int h = rem >> 6, hd = rem & 63;
                if (three == 0) v *= QSCALE;
                int bb = row >> 11, n = row & (Nn - 1);
                size_t off = (size_t)three * HS + ((size_t)(bb * Hh + h) * Nn + n) * HD + hd;
                obf[off] = f2bf(v);
            } else {
                of32[(size_t)row * Cc + col] = v + b0[col];
            }
        }
    }
}

// ---------------------------------------------------------------------------
// Flash attention over one (b,h), 64 query rows / block, 4 waves x 16 rows.
// qkvb layout: [3][b][h][n][hd] bf16; writes ctx [b][n][h*64+hd] bf16.
// ---------------------------------------------------------------------------
__global__ __launch_bounds__(128) void attn_kernel(const u16* __restrict__ qkvb,
                                                   u16* __restrict__ ctx) {
    __shared__ alignas(16) u16 qs[64][72];
    __shared__ alignas(16) u16 ks[32][72];
    __shared__ alignas(16) u16 vts[64][34];      // V transposed: [hd][key]
    __shared__ alignas(16) u16 ps[4][16][36];    // per-wave P staging

    const int tid = threadIdx.x, lane = tid & 31, wid = tid >> 5;
    const int lm = lane & 15, hi = lane >> 4;
    const int bh = blockIdx.y;                   // b*H + h
    const int n0 = blockIdx.x * 64;

    const u16* qp = qkvb + (size_t)bh * Nn * HD;
    const u16* kp = qkvb + HS + (size_t)bh * Nn * HD;
    const u16* vp = qkvb + 2 * HS + (size_t)bh * Nn * HD;

    // stage the 64x64 q tile once
    {
        int r = tid >> 1, sg = (tid & 1) * 32;
        const u16* g = qp + (size_t)(n0 + r) * HD + sg;
#if HAVE_ASYNC
        #pragma unroll
        for (int q = 0; q < 4; ++q)
            __builtin_amdgcn_global_load_async_to_lds_b128(AS1(g + 8 * q), AS3(&qs[r][sg + 8 * q]), 0, 0);
        ASYNC_WAIT();
#else
        const uint4* src = (const uint4*)g;
        *(uint4*)&qs[r][sg]      = src[0];
        *(uint4*)&qs[r][sg + 8]  = src[1];
        *(uint4*)&qs[r][sg + 16] = src[2];
        *(uint4*)&qs[r][sg + 24] = src[3];
#endif
    }
    __syncthreads();

    Frag qa[2];
    #pragma unroll
    for (int k2 = 0; k2 < 2; ++k2)
        #pragma unroll
        for (int j = 0; j < 8; ++j) {
            int ka = k2 * 32 + hi * 16 + 2 * j;
            qa[k2].u[j] = *(const u32*)&qs[wid * 16 + lm][ka];
        }

    float mrow[8], lrow[8];
    #pragma unroll
    for (int i = 0; i < 8; ++i) { mrow[i] = -3.0e38f; lrow[i] = 0.f; }
    v8f o0 = {}, o1 = {}, o2 = {}, o3 = {};

    for (int kb = 0; kb < Nn; kb += 32) {
        // stage 32x64 K chunk (async copy: no format change needed)
        {
            int r = tid >> 2, sg = (tid & 3) * 16;
            const u16* g = kp + (size_t)(kb + r) * HD + sg;
#if HAVE_ASYNC
            __builtin_amdgcn_global_load_async_to_lds_b128(AS1(g),     AS3(&ks[r][sg]),     0, 0);
            __builtin_amdgcn_global_load_async_to_lds_b128(AS1(g + 8), AS3(&ks[r][sg + 8]), 0, 0);
#else
            const uint4* src = (const uint4*)g;
            *(uint4*)&ks[r][sg]     = src[0];
            *(uint4*)&ks[r][sg + 8] = src[1];
#endif
        }
        // stage 32x64 V chunk transposed -> vts[hd][key]
        {
            int r = tid >> 2, sg = (tid & 3) * 16;
            const uint4* src = (const uint4*)(vp + (size_t)(kb + r) * HD + sg);
            uint4 d0 = src[0], d1 = src[1];
            u32 w[8] = {d0.x, d0.y, d0.z, d0.w, d1.x, d1.y, d1.z, d1.w};
            #pragma unroll
            for (int i = 0; i < 8; ++i) {
                vts[sg + 2 * i][r]     = (u16)(w[i] & 0xffffu);
                vts[sg + 2 * i + 1][r] = (u16)(w[i] >> 16);
            }
        }
        // prefetch next chunk while this one is consumed
        if (kb + 32 < Nn) {
            int r = tid >> 2, sg = (tid & 3) * 16;
            __builtin_prefetch(kp + (size_t)(kb + 32 + r) * HD + sg, 0, 3);
            __builtin_prefetch(vp + (size_t)(kb + 32 + r) * HD + sg, 0, 3);
        }
#if HAVE_ASYNC
        ASYNC_WAIT();
#endif
        __syncthreads();

        // S = q @ k^T : two 16x16 tiles (keys 0-15, 16-31), K=64 split in 2
        v8f s0 = {}, s1 = {};
        Frag bf;
        #pragma unroll
        for (int k2 = 0; k2 < 2; ++k2) {
            #pragma unroll
            for (int j = 0; j < 8; ++j)
                bf.u[j] = *(const u32*)&ks[lm][k2 * 32 + hi * 16 + 2 * j];
            s0 = __builtin_amdgcn_wmma_f32_16x16x32_bf16(false, qa[k2].v, false, bf.v, (short)0, s0, false, false);
            #pragma unroll
            for (int j = 0; j < 8; ++j)
                bf.u[j] = *(const u32*)&ks[16 + lm][k2 * 32 + hi * 16 + 2 * j];
            s1 = __builtin_amdgcn_wmma_f32_16x16x32_bf16(false, qa[k2].v, false, bf.v, (short)0, s1, false, false);
        }

        // online softmax; rows of the C-tile live across each 16-lane half
        #pragma unroll
        for (int i = 0; i < 8; ++i) {
            float a = s0[i], b = s1[i];
            float mx = rmax16(fmaxf(a, b));
            float mn = fmaxf(mrow[i], mx);
            float sc = exp2f((mrow[i] - mn) * LOG2E);
            float p0 = exp2f((a - mn) * LOG2E);
            float p1 = exp2f((b - mn) * LOG2E);
            lrow[i] = lrow[i] * sc + rsum16(p0 + p1);
            mrow[i] = mn;
            o0[i] *= sc; o1[i] *= sc; o2[i] *= sc; o3[i] *= sc;
            int r = i + hi * 8;
            ps[wid][r][lm]      = f2bf(p0);
            ps[wid][r][16 + lm] = f2bf(p1);
        }
        asm volatile("s_wait_dscnt 0x0" ::: "memory");  // wave-private LDS handoff

        // P (16x32) as A-fragment, then O += P @ V (4 hd tiles)
        Frag pa;
        #pragma unroll
        for (int j = 0; j < 8; ++j) {
            int kk = hi * 16 + 2 * j;
            pa.u[j] = *(const u32*)&ps[wid][lm][kk];
        }
        #pragma unroll
        for (int j = 0; j < 8; ++j) bf.u[j] = *(const u32*)&vts[lm][hi * 16 + 2 * j];
        o0 = __builtin_amdgcn_wmma_f32_16x16x32_bf16(false, pa.v, false, bf.v, (short)0, o0, false, false);
        #pragma unroll
        for (int j = 0; j < 8; ++j) bf.u[j] = *(const u32*)&vts[16 + lm][hi * 16 + 2 * j];
        o1 = __builtin_amdgcn_wmma_f32_16x16x32_bf16(false, pa.v, false, bf.v, (short)0, o1, false, false);
        #pragma unroll
        for (int j = 0; j < 8; ++j) bf.u[j] = *(const u32*)&vts[32 + lm][hi * 16 + 2 * j];
        o2 = __builtin_amdgcn_wmma_f32_16x16x32_bf16(false, pa.v, false, bf.v, (short)0, o2, false, false);
        #pragma unroll
        for (int j = 0; j < 8; ++j) bf.u[j] = *(const u32*)&vts[48 + lm][hi * 16 + 2 * j];
        o3 = __builtin_amdgcn_wmma_f32_16x16x32_bf16(false, pa.v, false, bf.v, (short)0, o3, false, false);

        __syncthreads();
    }

    // finalize: O /= l, write ctx bf16 [b][n][h*64+hd]
    int b = bh / Hh, h = bh - b * Hh;
    #pragma unroll
    for (int i = 0; i < 8; ++i) {
        float inv = 1.0f / lrow[i];
        int n = n0 + wid * 16 + i + hi * 8;
        size_t base = ((size_t)(b * Nn + n)) * Cc + h * HD + lm;
        ctx[base]      = f2bf(o0[i] * inv);
        ctx[base + 16] = f2bf(o1[i] * inv);
        ctx[base + 32] = f2bf(o2[i] * inv);
        ctx[base + 48] = f2bf(o3[i] * inv);
    }
}

// ---------------------------------------------------------------------------
// out[row][col] += SCALING * sum_j t2[row][j] * BP[col][j]
// ---------------------------------------------------------------------------
__global__ __launch_bounds__(256) void lora_add_kernel(float* __restrict__ out,
                                                       const float* __restrict__ T2,
                                                       const float* __restrict__ BP) {
    int idx = blockIdx.x * 256 + threadIdx.x;   // 8192*768
    int row = idx / Cc;
    int col = idx - row * Cc;
    float s = 0.f;
    #pragma unroll
    for (int j = 0; j < 8; ++j) s += T2[row * 8 + j] * BP[col * 8 + j];
    out[idx] += SCALING * s;
}

extern "C" void kernel_launch(void* const* d_in, const int* in_sizes, int n_in,
                              void* d_out, int out_size, void* d_ws, size_t ws_size,
                              hipStream_t stream) {
    (void)in_sizes; (void)n_in; (void)out_size; (void)ws_size;
    const float* x   = (const float*)d_in[0];
    const float* wq  = (const float*)d_in[1];
    const float* qb  = (const float*)d_in[2];
    const float* vb  = (const float*)d_in[3];
    const float* laq = (const float*)d_in[4];
    const float* lbq = (const float*)d_in[5];
    const float* wp  = (const float*)d_in[6];
    const float* pb  = (const float*)d_in[7];
    const float* lap = (const float*)d_in[8];
    const float* lbp = (const float*)d_in[9];
    float* out = (float*)d_out;

    // workspace: t1(256K) | t2(256K) | qkvb bf16 (37.75M) | ctx bf16 (12.6M)
    char* ws = (char*)d_ws;
    float* t1   = (float*)ws;
    float* t2   = (float*)(ws + (size_t)262144);
    u16*   qkvb = (u16*)(ws + (size_t)524288);
    u16*   ctx  = (u16*)(ws + (size_t)524288 + 3 * HS * sizeof(u16));

    lora_t_kernel<<<256, 256, 0, stream>>>(x, laq, t1);
    gemm_kernel<true, 0><<<dim3(36, 64), 256, 0, stream>>>(
        x, wq, qb, vb, t1, lbq, qkvb, nullptr);
    attn_kernel<<<dim3(32, 48), 128, 0, stream>>>(qkvb, ctx);
    gemm_kernel<false, 1><<<dim3(12, 64), 256, 0, stream>>>(
        ctx, wp, pb, nullptr, nullptr, nullptr, nullptr, out);
    lora_t_kernel<<<256, 256, 0, stream>>>(out, lap, t2);
    lora_add_kernel<<<24576, 256, 0, stream>>>(out, t2, lbp);
}
